// Graph_Attention_91259465105657
// MI455X (gfx1250) — matrix operand order repaired and checked by reading, not compiled, not run on hardware
//
#include <hip/hip_runtime.h>

// ---------------------------------------------------------------------------
// Sizes
// ---------------------------------------------------------------------------
#define NNODE   100
#define BATCH   4
#define TSTEPS  32
#define BT      128            // BATCH*TSTEPS
#define NROWS   12800          // BT*NNODE
#define HID     64
#define NEDGEP  10000          // NNODE*NNODE (positions per block)
#define NEDGES  1280000        // BT*NEDGEP
#define NGROUP  625            // NEDGEP/16
#define STATES  10
#define EPS_BN  1e-5f
#define BLKF    (NNODE * HID)  // 6400 floats per (bt) u/v block

typedef __attribute__((ext_vector_type(16))) __bf16    v16bf;
typedef __attribute__((ext_vector_type(8)))  float     v8f;
typedef __attribute__((ext_vector_type(8)))  unsigned  v8u;

#if __has_builtin(__builtin_amdgcn_tensor_load_to_lds) && __has_builtin(__builtin_amdgcn_s_wait_tensorcnt)
#define USE_TDM 1
#endif

__device__ __forceinline__ float eluf(float x) {
    return x > 0.f ? x : (__expf(x) - 1.f);
}
__device__ __forceinline__ unsigned bf16r(float x) {
    unsigned u = __float_as_uint(x);
    return (u + 0x7FFFu + ((u >> 16) & 1u)) >> 16;
}
__device__ __forceinline__ unsigned packbf(float lo, float hi) {
#if __has_builtin(__builtin_amdgcn_cvt_pk_bf16_f32)
    auto t = __builtin_amdgcn_cvt_pk_bf16_f32(lo, hi);   // v_cvt_pk_bf16_f32
    return __builtin_bit_cast(unsigned, t);
#else
    return bf16r(lo) | (bf16r(hi) << 16);
#endif
}

#if defined(USE_TDM)
// ---------------------------------------------------------------------------
// Tensor Data Mover: 1-D tile (nElem f32) global -> LDS.  D# per ISA §8.3/8.4.
// group0: count=1 | lds_addr | global_addr[56:0] | type=2
// group1: data_size=4B, tensor_dim0=tile_dim0=nElem, tensor_dim1=tile_dim1=1
// 6-arg builtin form (clang-23 / therock): (v4u, v8i, v4i, v4i, v8i, cpol)
// ---------------------------------------------------------------------------
typedef unsigned t_v4u __attribute__((ext_vector_type(4)));
typedef int      t_v8i __attribute__((ext_vector_type(8)));
typedef int      t_v4i __attribute__((ext_vector_type(4)));

__device__ __forceinline__ void tdm_load_to_lds(const float* gsrc, float* lds_dst, int nElem)
{
    unsigned long long ga = (unsigned long long)gsrc;
    unsigned lds = (unsigned)(unsigned long long)lds_dst;   // low 32 bits = LDS offset
    t_v4u g0;
    g0[0] = 1u;                                             // count=1
    g0[1] = lds;
    g0[2] = (unsigned)(ga & 0xFFFFFFFFull);
    g0[3] = (unsigned)((ga >> 32) & 0x01FFFFFFull) | (2u << 30);  // type=2 ("image")
    t_v8i g1;
    g1[0] = (int)(2u << 16);                                // data_size = 4 bytes
    g1[1] = (int)(((unsigned)nElem & 0xFFFFu) << 16);       // tensor_dim0[15:0]
    g1[2] = (int)((((unsigned)nElem >> 16) & 0xFFFFu) | (1u << 16)); // td0 hi | tensor_dim1=1
    g1[3] = (int)(((unsigned)nElem & 0xFFFFu) << 16);       // tile_dim0
    g1[4] = 1;                                              // tile_dim1 = 1
    g1[5] = nElem;                                          // tensor_dim0_stride lo32
    g1[6] = (int)(((unsigned)nElem & 0xFFFFu) << 16);       // stride0 hi=0 | stride1 lo16
    g1[7] = 0;
    t_v4i z4 = {0, 0, 0, 0};
    t_v8i z8 = {0, 0, 0, 0, 0, 0, 0, 0};
    __builtin_amdgcn_tensor_load_to_lds(g0, g1, z4, z4, z8, 0);
}
#endif

// ---------------------------------------------------------------------------
// Workspace layout (float offsets)
// ---------------------------------------------------------------------------
#define OFF_MEAN  0u            // 640000  sum over bt of h2 per (pos,f)
#define OFF_Q     640000u       // 64      sum of h2^2 over all E rows
#define OFF_S     640064u       // 64      column sum of meanAcc
#define OFF_SSQ   640128u       // 64      (dummy)
#define OFF_ESUM  640192u
#define OFF_ESQ   640256u
#define OFF_GSUM  640320u
#define OFF_GSQ   640384u
#define OFF_OSUM  640448u
#define OFF_OSQ   640512u
#define OFF_AE    640576u
#define OFF_BE_   640640u
#define OFF_AG    640704u
#define OFF_BG    640768u
#define OFF_AO    640832u
#define OFF_BO    640896u
#define OFF_A2    640960u
#define OFF_B2    641024u
#define OFF_W2PK  641088u       // 2048 uint32 (bf16 B fragments of att_mlp.w2)
#define OFF_XR    643136u       // 25600
#define OFF_FEAT  668736u       // 819200 (pre-BN, then BN'd in place)
#define OFF_U     1487936u      // 819200 (u; reused as agg)
#define OFF_V     2307136u      // 819200 (v; reused as gconv pre-BN)
#define OFF_H2G   3126336u      // 819200
#define OFF_OPRE  3945536u      // 8192

// ---------------------------------------------------------------------------
// trivial kernels
// ---------------------------------------------------------------------------
__global__ void k_zero(float* p, int n) {
    int i = blockIdx.x * 256 + threadIdx.x;
    if (i < n) p[i] = 0.f;
}

// x (B*n, T, 2) -> xr (B*T*n, 2)
__global__ void k_xr(const float* __restrict__ x, float* __restrict__ xr) {
    int idx = blockIdx.x * 256 + threadIdx.x;
    if (idx >= NROWS * 2) return;
    int c = idx & 1;
    int rest = idx >> 1;
    int k = rest % NNODE;
    int bt = rest / NNODE;
    int b = bt >> 5;        // /32
    int t = bt & 31;
    xr[idx] = x[(((b * NNODE + k) * TSTEPS) + t) * 2 + c];
}

// ---------------------------------------------------------------------------
// enc MLP (pre-BN) + att_enc MLP + u/v projections.  4 rows / 256-thread block.
// ---------------------------------------------------------------------------
__global__ __launch_bounds__(256) void k_encatt(
    const float* __restrict__ xr,
    const float* __restrict__ ew1, const float* __restrict__ eb1,
    const float* __restrict__ ew2, const float* __restrict__ eb2,
    const float* __restrict__ aw1, const float* __restrict__ ab1,
    const float* __restrict__ aw2, const float* __restrict__ ab2,
    const float* __restrict__ mw1, const float* __restrict__ mb1,
    float* __restrict__ featPre, float* __restrict__ uB, float* __restrict__ vB)
{
    __shared__ float h1e[4][HID], h1a[4][HID], xeS[4][HID];
    int rloc = threadIdx.x >> 6, f = threadIdx.x & 63;
    int row = blockIdx.x * 4 + rloc;
    float x0 = xr[row * 2], x1 = xr[row * 2 + 1];
    h1e[rloc][f] = eluf(x0 * ew1[f] + x1 * ew1[HID + f] + eb1[f]);
    h1a[rloc][f] = eluf(x0 * aw1[f] + x1 * aw1[HID + f] + ab1[f]);
    __syncthreads();
    float a1 = eb2[f], a2 = ab2[f];
    #pragma unroll 8
    for (int k = 0; k < HID; ++k) {
        a1 += h1e[rloc][k] * ew2[k * HID + f];
        a2 += h1a[rloc][k] * aw2[k * HID + f];
    }
    featPre[row * HID + f] = eluf(a1);
    xeS[rloc][f] = eluf(a2);
    __syncthreads();
    float au = mb1[f], av = 0.f;
    #pragma unroll 8
    for (int k = 0; k < HID; ++k) {
        float xv = xeS[rloc][k];
        au += xv * mw1[k * HID + f];
        av += xv * mw1[(k + HID) * HID + f];
    }
    uB[row * HID + f] = au;
    vB[row * HID + f] = av;
}

// ---------------------------------------------------------------------------
// column statistics: sum / sum-of-squares over rows of a (nrows x 64) matrix
// ---------------------------------------------------------------------------
__global__ __launch_bounds__(256) void k_colstats(
    const float* __restrict__ src, int nrows,
    float* __restrict__ sum, float* __restrict__ sq)
{
    __shared__ float s1[256], s2[256];
    int col = blockIdx.x;
    float a = 0.f, b = 0.f;
    for (int r = threadIdx.x; r < nrows; r += 256) {
        float v = src[r * HID + col];
        a += v; b += v * v;
    }
    s1[threadIdx.x] = a; s2[threadIdx.x] = b;
    __syncthreads();
    for (int off = 128; off > 0; off >>= 1) {
        if (threadIdx.x < (unsigned)off) {
            s1[threadIdx.x] += s1[threadIdx.x + off];
            s2[threadIdx.x] += s2[threadIdx.x + off];
        }
        __syncthreads();
    }
    if (threadIdx.x == 0) { sum[col] = s1[0]; sq[col] = s2[0]; }
}

__global__ void k_bnfin(const float* sum, const float* sq,
                        const float* g, const float* be, float invN,
                        float* a_out, float* b_out)
{
    int f = threadIdx.x;
    float m = sum[f] * invN;
    float var = sq[f] * invN - m * m;
    float a = g[f] * rsqrtf(var + EPS_BN);
    a_out[f] = a;
    b_out[f] = be[f] - m * a;
}

__global__ void k_applybn(float* p, const float* a, const float* b, int n) {
    int i = blockIdx.x * 256 + threadIdx.x;
    if (i >= n) return;
    int f = i & 63;
    p[i] = p[i] * a[f] + b[f];
}

// ---------------------------------------------------------------------------
// pack att_mlp.w2 (64x64, row-major [k][n]) into bf16 WMMA B fragments.
// B (32x16), contiguous k order as A uses: lanes 0-15 carry N=lane, K=0..15 (VGPR r = K 2r,2r+1
// in [15:0],[31:16]); lanes 16-31 carry N=lane-16, K=16..31.
// ---------------------------------------------------------------------------
__global__ void k_w2pack(const float* __restrict__ w2, unsigned* __restrict__ w2pk) {
    int idx = blockIdx.x * 256 + threadIdx.x;     // 0..2047
    if (idx >= 2048) return;
    int r    = idx & 7;
    int lane = (idx >> 3) & 31;
    int kt   = (idx >> 8) & 1;
    int t    = idx >> 9;
    int n    = t * 16 + (lane & 15);
    int kloc = (lane < 16) ? (2 * r) : (16 + 2 * r);
    int k    = kt * 32 + kloc;
    w2pk[idx] = packbf(w2[k * HID + n], w2[(k + 1) * HID + n]);
}

// ---------------------------------------------------------------------------
// THE hot kernel: per-edge  h2 = ELU( ELU(u_i + v_j) @ W2 + b2 )
// accumulated into meanAcc (sum over bt) and Qacc (sum of squares over all E).
// grid = (125, 8), block = 160 (5 waves, all with full EXEC for WMMA).
// Each wave owns one group of 16 edge positions and all 64 output features
// (4 N-tiles x 2 K-halves = 8 WMMAs per bt), looping over 16 bt blocks.
// u/v blocks are staged into LDS by the Tensor Data Mover (double-buffered,
// overlapped with compute) when available; synchronous copy + prefetch else.
// ---------------------------------------------------------------------------
__global__ __launch_bounds__(160) void k_edge(
    const float* __restrict__ uB, const float* __restrict__ vB,
    const unsigned* __restrict__ w2pk, const float* __restrict__ b2,
    float* __restrict__ meanAcc, float* __restrict__ Qacc)
{
#if defined(USE_TDM)
    __shared__ float su[2][BLKF];   // 2 x 25.6 KB
    __shared__ float sv[2][BLKF];
#else
    __shared__ float su[1][BLKF];
    __shared__ float sv[1][BLKF];
#endif

    const int tid  = threadIdx.x;
    const int lane = tid & 31;
    const int wid  = tid >> 5;
    const int group = blockIdx.x * 5 + wid;          // 0..624, always valid
    const int half = lane >> 4;

    // A-matrix row for this lane (edge position within the group)
    const int eA = group * 16 + (lane & 15);
    const int iA = eA / NNODE;
    const int jA = eA - iA * NNODE;

    // preload B fragments (att_mlp W2 in bf16), 4 N-tiles x 2 K-halves
    v16bf Bf[4][2];
    #pragma unroll
    for (int t = 0; t < 4; ++t)
        #pragma unroll
        for (int kt = 0; kt < 2; ++kt) {
            v8u bu;
            const unsigned* p = w2pk + (((t * 2 + kt) * 32 + lane) << 3);
            #pragma unroll
            for (int r = 0; r < 8; ++r) bu[r] = p[r];
            Bf[t][kt] = __builtin_bit_cast(v16bf, bu);
        }
    float b2v[4];
    #pragma unroll
    for (int t = 0; t < 4; ++t) b2v[t] = b2[t * 16 + (lane & 15)];

    float sacc[4][8];
    float qacc[4] = {0.f, 0.f, 0.f, 0.f};
    #pragma unroll
    for (int t = 0; t < 4; ++t)
        #pragma unroll
        for (int r = 0; r < 8; ++r) sacc[t][r] = 0.f;

#if defined(USE_TDM)
    // prologue DMA: buffer 0 <- bt(c=0)
    if (wid == 0) {
        const int bt0 = blockIdx.y * 16;
        tdm_load_to_lds(uB + (size_t)bt0 * BLKF, &su[0][0], BLKF);
        tdm_load_to_lds(vB + (size_t)bt0 * BLKF, &sv[0][0], BLKF);
    }
#endif

    for (int c = 0; c < 16; ++c) {
        const int bt = blockIdx.y * 16 + c;
#if defined(USE_TDM)
        const int p = c & 1;
        if (wid == 0) __builtin_amdgcn_s_wait_tensorcnt(0);
        __syncthreads();                        // buffer p is ready for everyone
        if (wid == 0 && c + 1 < 16) {           // kick DMA for next bt into p^1
            tdm_load_to_lds(uB + (size_t)(bt + 1) * BLKF, &su[p ^ 1][0], BLKF);
            tdm_load_to_lds(vB + (size_t)(bt + 1) * BLKF, &sv[p ^ 1][0], BLKF);
        }
        const float* ur = &su[p][0] + iA * HID;
        const float* vr = &sv[p][0] + jA * HID;
#else
        __syncthreads();
        {
            const float4* ug4 = reinterpret_cast<const float4*>(uB + (size_t)bt * BLKF);
            const float4* vg4 = reinterpret_cast<const float4*>(vB + (size_t)bt * BLKF);
            float4* su4 = reinterpret_cast<float4*>(&su[0][0]);
            float4* sv4 = reinterpret_cast<float4*>(&sv[0][0]);
            for (int idx = tid; idx < BLKF / 4; idx += 160) {
                su4[idx] = ug4[idx];
                sv4[idx] = vg4[idx];
            }
        }
        __syncthreads();
        if (c + 1 < 16) {                       // global_prefetch next block
            const float* un = uB + (size_t)(bt + 1) * BLKF;
            const float* vn = vB + (size_t)(bt + 1) * BLKF;
            for (int idx = tid * 32; idx < BLKF; idx += 160 * 32) {
                __builtin_prefetch(un + idx, 0, 0);
                __builtin_prefetch(vn + idx, 0, 0);
            }
        }
        const float* ur = &su[0][0] + iA * HID;
        const float* vr = &sv[0][0] + jA * HID;
#endif

        // build A fragments: h1 = ELU(u_i + v_j) in bf16
        // A in contiguous k order (as packed B): lanes0-15 M=lane, K pairs V0..7 = 0..15
        //                  lanes16-31 M=lane-16, V0..7 = 16..31
        v8u a0u, a1u;
        #pragma unroll
        for (int r = 0; r < 8; ++r) {
            int pp = half * 8 + r;
            int k0 = pp * 2;
            a0u[r] = packbf(eluf(ur[k0] + vr[k0]), eluf(ur[k0 + 1] + vr[k0 + 1]));
            int k1 = 32 + pp * 2;
            a1u[r] = packbf(eluf(ur[k1] + vr[k1]), eluf(ur[k1 + 1] + vr[k1 + 1]));
        }
        v16bf A0 = __builtin_bit_cast(v16bf, a0u);
        v16bf A1 = __builtin_bit_cast(v16bf, a1u);

        #pragma unroll
        for (int t = 0; t < 4; ++t) {
            v8f cc = {};
            cc = __builtin_amdgcn_wmma_f32_16x16x32_bf16(false, A0, false, Bf[t][0],
                                                         (short)0, cc, false, false);
            cc = __builtin_amdgcn_wmma_f32_16x16x32_bf16(false, A1, false, Bf[t][1],
                                                         (short)0, cc, false, false);
            #pragma unroll
            for (int r = 0; r < 8; ++r) {
                float h2 = eluf(cc[r] + b2v[t]);
                sacc[t][r] += h2;
                qacc[t] += h2 * h2;
            }
        }
    }

    // C/D layout: lanes0-15 -> M=r, lanes16-31 -> M=r+8; N = lane&15
    const int mCoff = (lane < 16) ? 0 : 8;
    const int nC = lane & 15;
    #pragma unroll
    for (int t = 0; t < 4; ++t) {
        #pragma unroll
        for (int r = 0; r < 8; ++r) {
            int e = group * 16 + mCoff + r;
            atomicAdd(&meanAcc[e * HID + t * 16 + nC], sacc[t][r]);
        }
        atomicAdd(&Qacc[t * 16 + nC], qacc[t]);
    }
}

// ---------------------------------------------------------------------------
// attention coefficients + per-position attention weights
// ---------------------------------------------------------------------------
__global__ void k_attcoef(const float* S, const float* Q,
                          const float* g, const float* be,
                          float* A2, float* B2)
{
    int f = threadIdx.x;
    const float invE = 1.f / (float)NEDGES;
    float m = S[f] * invE;
    float var = Q[f] * invE - m * m;
    float a = g[f] * rsqrtf(var + EPS_BN);
    A2[f] = a * (1.f / (float)BT);   // meanAcc/128 folded in
    B2[f] = be[f] - m * a;
}

__global__ __launch_bounds__(256) void k_att(
    const float* __restrict__ meanAcc, const float* A2, const float* B2,
    const float* fw, const float* fb, float* __restrict__ attOut)
{
    __shared__ float sA[HID], sB[HID], sW[2 * HID];
    int t = threadIdx.x;
    if (t < HID) { sA[t] = A2[t]; sB[t] = B2[t]; }
    if (t < 2 * HID) sW[t] = fw[t];
    __syncthreads();
    int pos = blockIdx.x * 256 + t;
    if (pos >= NEDGEP) return;
    float z0 = fb[0], z1 = fb[1];
    #pragma unroll 8
    for (int f = 0; f < HID; ++f) {
        float v = meanAcc[pos * HID + f] * sA[f] + sB[f];
        z0 += v * sW[f * 2];
        z1 += v * sW[f * 2 + 1];
    }
    attOut[pos] = 1.f / (1.f + __expf(z0 - z1));   // softmax(.)[1]
}

// ---------------------------------------------------------------------------
// weighted aggregation: agg[bt,j,f] = sum_i feat[bt,i,f] * att[i*n+j]
// ---------------------------------------------------------------------------
__global__ __launch_bounds__(64) void k_agg(
    const float* __restrict__ feat, const float* __restrict__ att,
    float* __restrict__ agg)
{
    __shared__ float satt[NNODE];
    int bj = blockIdx.x;                // bt*100 + j
    int bt = bj / NNODE;
    int j  = bj - bt * NNODE;
    int f  = threadIdx.x;
    for (int i = f; i < NNODE; i += 64) satt[i] = att[i * NNODE + j];
    __syncthreads();
    float acc = 0.f;
    #pragma unroll 4
    for (int i = 0; i < NNODE; ++i)
        acc += feat[(bt * NNODE + i) * HID + f] * satt[i];
    agg[bj * HID + f] = acc;
}

// ---------------------------------------------------------------------------
// gconv MLP (two 64->64 ELU layers, pre-BN out) ; then BN + fc
// ---------------------------------------------------------------------------
__global__ __launch_bounds__(256) void k_gmlp(
    const float* __restrict__ agg,
    const float* __restrict__ w1, const float* __restrict__ b1,
    const float* __restrict__ w2, const float* __restrict__ b2,
    float* __restrict__ gpre)
{
    __shared__ float sIn[4][HID], sH[4][HID];
    int rloc = threadIdx.x >> 6, f = threadIdx.x & 63;
    int row = blockIdx.x * 4 + rloc;
    sIn[rloc][f] = agg[row * HID + f];
    __syncthreads();
    float a = b1[f];
    #pragma unroll 8
    for (int k = 0; k < HID; ++k) a += sIn[rloc][k] * w1[k * HID + f];
    sH[rloc][f] = eluf(a);
    __syncthreads();
    float a2 = b2[f];
    #pragma unroll 8
    for (int k = 0; k < HID; ++k) a2 += sH[rloc][k] * w2[k * HID + f];
    gpre[row * HID + f] = eluf(a2);
}

__global__ __launch_bounds__(256) void k_gfc(
    const float* __restrict__ gpre, const float* aG, const float* bG,
    const float* __restrict__ fcw, const float* __restrict__ fcb,
    float* __restrict__ h2g)
{
    __shared__ float sB[4][HID];
    int rloc = threadIdx.x >> 6, f = threadIdx.x & 63;
    int row = blockIdx.x * 4 + rloc;
    sB[rloc][f] = gpre[row * HID + f] * aG[f] + bG[f];
    __syncthreads();
    float a = fcb[f];
    #pragma unroll 8
    for (int k = 0; k < HID; ++k) a += sB[rloc][k] * fcw[k * HID + f];
    h2g[row * HID + f] = a;
}

// ---------------------------------------------------------------------------
// out_mlp: hid row (12800-wide interleave of feat/h2g) -> 64 -> 64 (pre-BN)
// ---------------------------------------------------------------------------
__global__ __launch_bounds__(256) void k_out(
    const float* __restrict__ feat, const float* __restrict__ h2g,
    const float* __restrict__ ow1, const float* __restrict__ ob1,
    const float* __restrict__ ow2, const float* __restrict__ ob2,
    float* __restrict__ outPre)
{
    __shared__ float hid[2 * NNODE * HID];   // 12800 floats
    __shared__ float pbuf[256];
    __shared__ float o1[HID];
    int row = blockIdx.x, t = threadIdx.x;
    for (int idx = t; idx < 2 * NNODE * HID; idx += 256) {
        int s = idx & 1;
        int f = (idx >> 1) & 63;
        int k = idx >> 7;
        const float* src = s ? h2g : feat;
        hid[idx] = src[(row * NNODE + k) * HID + f];
    }
    __syncthreads();
    int h = t & 63, part = t >> 6;
    float acc = 0.f;
    int base = part * 3200;
    for (int idx = base; idx < base + 3200; ++idx)
        acc += hid[idx] * ow1[idx * HID + h];
    pbuf[t] = acc;
    __syncthreads();
    if (t < HID) {
        float o = pbuf[t] + pbuf[t + 64] + pbuf[t + 128] + pbuf[t + 192] + ob1[t];
        o1[t] = eluf(o);
    }
    __syncthreads();
    if (t < HID) {
        float a2 = ob2[t];
        #pragma unroll 8
        for (int k = 0; k < HID; ++k) a2 += o1[k] * ow2[k * HID + t];
        outPre[row * HID + t] = eluf(a2);
    }
}

// final BN + fc (64->10) + softmax
__global__ __launch_bounds__(128) void k_fc(
    const float* __restrict__ outPre, const float* aO, const float* bO,
    const float* __restrict__ fcw, const float* __restrict__ fcb,
    float* __restrict__ out)
{
    int row = threadIdx.x;     // 128 rows
    float lg[STATES];
    #pragma unroll
    for (int c = 0; c < STATES; ++c) lg[c] = fcb[c];
    for (int k = 0; k < HID; ++k) {
        float v = outPre[row * HID + k] * aO[k] + bO[k];
        #pragma unroll
        for (int c = 0; c < STATES; ++c) lg[c] += v * fcw[k * STATES + c];
    }
    float mx = lg[0];
    #pragma unroll
    for (int c = 1; c < STATES; ++c) mx = fmaxf(mx, lg[c]);
    float se = 0.f;
    #pragma unroll
    for (int c = 0; c < STATES; ++c) { lg[c] = __expf(lg[c] - mx); se += lg[c]; }
    float inv = 1.f / se;
    #pragma unroll
    for (int c = 0; c < STATES; ++c) out[row * STATES + c] = lg[c] * inv;
}

// ---------------------------------------------------------------------------
// host launcher
// ---------------------------------------------------------------------------
struct Ptrs {
    const float *ew1, *eb1, *ew2, *eb2, *eg, *ebe;        // enc
    const float *aew1, *aeb1, *aew2, *aeb2;               // att_enc
    const float *mw1, *mb1, *mw2, *mb2, *mg, *mbe;        // att_mlp
    const float *afcw, *afcb;                              // att_fc
    const float *gw1, *gb1, *gw2, *gb2, *gg, *gbe;        // gconv_mlp
    const float *gfcw, *gfcb;                              // gconv_fc
    const float *ow1, *ob1, *ow2, *ob2, *og, *obe;        // out_mlp
    const float *fcw, *fcb;                                // fc
    const float *x;
};

extern "C" void kernel_launch(void* const* d_in, const int* in_sizes, int n_in,
                              void* d_out, int out_size, void* d_ws, size_t ws_size,
                              hipStream_t stream) {
    (void)n_in; (void)out_size; (void)ws_size;
    auto f = [&](int i) { return (const float*)d_in[i]; };
    Ptrs P;
    if (in_sizes[0] == NROWS * 2) {
        // insertion-order flatten: x, edge_index, params(dicts in insertion order), timesteps
        P.x = f(0);
        P.ew1 = f(2);  P.eb1 = f(3);  P.ew2 = f(4);  P.eb2 = f(5);  P.eg = f(6);  P.ebe = f(7);
        P.aew1 = f(8); P.aeb1 = f(9); P.aew2 = f(10); P.aeb2 = f(11);
        P.mw1 = f(12); P.mb1 = f(13); P.mw2 = f(14); P.mb2 = f(15); P.mg = f(16); P.mbe = f(17);
        P.afcw = f(18); P.afcb = f(19);
        P.gw1 = f(20); P.gb1 = f(21); P.gw2 = f(22); P.gb2 = f(23); P.gg = f(24); P.gbe = f(25);
        P.gfcw = f(26); P.gfcb = f(27);
        P.ow1 = f(28); P.ob1 = f(29); P.ow2 = f(30); P.ob2 = f(31); P.og = f(32); P.obe = f(33);
        P.fcw = f(34); P.fcb = f(35);
    } else {
        // jax pytree (alphabetically sorted keys, recursive)
        P.aew1 = f(3); P.aeb1 = f(1); P.aew2 = f(4); P.aeb2 = f(2);
        P.afcb = f(5); P.afcw = f(6);
        P.mb1 = f(7); P.mb2 = f(8); P.mbe = f(9); P.mg = f(10); P.mw1 = f(11); P.mw2 = f(12);
        P.eb1 = f(13); P.eb2 = f(14); P.ebe = f(15); P.eg = f(16); P.ew1 = f(17); P.ew2 = f(18);
        P.fcb = f(19); P.fcw = f(20);
        P.gfcb = f(21); P.gfcw = f(22);
        P.gb1 = f(23); P.gb2 = f(24); P.gbe = f(25); P.gg = f(26); P.gw1 = f(27); P.gw2 = f(28);
        P.ob1 = f(29); P.ob2 = f(30); P.obe = f(31); P.og = f(32); P.ow1 = f(33); P.ow2 = f(34);
        P.x = f(36);
    }

    float* W = (float*)d_ws;
    float* out = (float*)d_out;
    float* attOut = out + BT * STATES;         // 10000 attention weights

    // zero atomic accumulators (meanAcc + Qacc)
    k_zero<<<(640064 + 255) / 256, 256, 0, stream>>>(W + OFF_MEAN, 640064);
    k_xr<<<100, 256, 0, stream>>>(P.x, W + OFF_XR);

    k_encatt<<<NROWS / 4, 256, 0, stream>>>(W + OFF_XR,
        P.ew1, P.eb1, P.ew2, P.eb2, P.aew1, P.aeb1, P.aew2, P.aeb2,
        P.mw1, P.mb1, W + OFF_FEAT, W + OFF_U, W + OFF_V);

    k_colstats<<<64, 256, 0, stream>>>(W + OFF_FEAT, NROWS, W + OFF_ESUM, W + OFF_ESQ);
    k_bnfin<<<1, 64, 0, stream>>>(W + OFF_ESUM, W + OFF_ESQ, P.eg, P.ebe,
                                  1.f / (float)NROWS, W + OFF_AE, W + OFF_BE_);
    k_applybn<<<(NROWS * HID) / 256, 256, 0, stream>>>(W + OFF_FEAT, W + OFF_AE, W + OFF_BE_,
                                                       NROWS * HID);

    k_w2pack<<<8, 256, 0, stream>>>(P.mw2, (unsigned*)(W + OFF_W2PK));

    dim3 eg(NGROUP / 5, 8);
    k_edge<<<eg, 160, 0, stream>>>(W + OFF_U, W + OFF_V,
                                   (const unsigned*)(W + OFF_W2PK), P.mb2,
                                   W + OFF_MEAN, W + OFF_Q);

    k_colstats<<<64, 256, 0, stream>>>(W + OFF_MEAN, NEDGEP, W + OFF_S, W + OFF_SSQ);
    k_attcoef<<<1, 64, 0, stream>>>(W + OFF_S, W + OFF_Q, P.mg, P.mbe,
                                    W + OFF_A2, W + OFF_B2);
    k_att<<<(NEDGEP + 255) / 256, 256, 0, stream>>>(W + OFF_MEAN, W + OFF_A2, W + OFF_B2,
                                                    P.afcw, P.afcb, attOut);

    k_agg<<<NROWS, 64, 0, stream>>>(W + OFF_FEAT, attOut, W + OFF_U);   // agg -> U buf

    k_gmlp<<<NROWS / 4, 256, 0, stream>>>(W + OFF_U, P.gw1, P.gb1, P.gw2, P.gb2,
                                          W + OFF_V);                   // gpre -> V buf
    k_colstats<<<64, 256, 0, stream>>>(W + OFF_V, NROWS, W + OFF_GSUM, W + OFF_GSQ);
    k_bnfin<<<1, 64, 0, stream>>>(W + OFF_GSUM, W + OFF_GSQ, P.gg, P.gbe,
                                  1.f / (float)NROWS, W + OFF_AG, W + OFF_BG);
    k_gfc<<<NROWS / 4, 256, 0, stream>>>(W + OFF_V, W + OFF_AG, W + OFF_BG,
                                         P.gfcw, P.gfcb, W + OFF_H2G);

    k_out<<<BT, 256, 0, stream>>>(W + OFF_FEAT, W + OFF_H2G,
                                  P.ow1, P.ob1, P.ow2, P.ob2, W + OFF_OPRE);
    k_colstats<<<64, 256, 0, stream>>>(W + OFF_OPRE, BT, W + OFF_OSUM, W + OFF_OSQ);
    k_bnfin<<<1, 64, 0, stream>>>(W + OFF_OSUM, W + OFF_OSQ, P.og, P.obe,
                                  1.f / (float)BT, W + OFF_AO, W + OFF_BO);
    k_fc<<<1, BT, 0, stream>>>(W + OFF_OPRE, W + OFF_AO, W + OFF_BO, P.fcw, P.fcb, out);
}
